// CIN_56392920597049
// MI455X (gfx1250) — matrix operand order repaired and checked by reading, not compiled, not run on hardware
//
#include <hip/hip_runtime.h>
#include <hip/hip_fp16.h>

typedef __attribute__((ext_vector_type(16))) _Float16 v16h;
typedef __attribute__((ext_vector_type(8)))  _Float16 v8h;
typedef __attribute__((ext_vector_type(8)))  float    v8f;
typedef int v4i_b128 __attribute__((vector_size(16)));   // matches builtin param

#define BATCH 512
#define NF    39      // F: number of fields
#define EMB   32      // D: embedding dim
#define UOUT  256     // units per layer
#define MROWS (BATCH*EMB)   // 16384 GEMM rows (b,d)
#define HKP0  64            // layer-0 Hk (=39) padded to 64
#define OUTW  768           // 3*256 concatenated output
#define MTILE 128           // rows per workgroup (4 batch elements)
#define NTILE 128           // cols per workgroup
#define WSTEP 4096          // f16 elems per 8KB W step tile

#if defined(__gfx1250__) && __has_builtin(__builtin_amdgcn_global_load_async_to_lds_b128)
#define ASYNC_LDS_COPY 1
#endif

// ---------------------------------------------------------------------------
// Prep 1: transpose inputs [B, F=39, D=32] f32 -> X0T [MROWS, 64] f16
// ---------------------------------------------------------------------------
__global__ void prep_x0(const float* __restrict__ in, _Float16* __restrict__ x0t) {
    int idx = blockIdx.x * blockDim.x + threadIdx.x;   // over MROWS*HKP0
    int j = idx & (HKP0 - 1);
    int m = idx >> 6;
    int b = m >> 5, d = m & 31;
    _Float16 v = (_Float16)0.0f;
    if (j < NF) v = (_Float16)in[(b * NF + j) * EMB + d];
    x0t[idx] = v;
}

// ---------------------------------------------------------------------------
// Prep 2: convert W f32 [F*Hk, 256] -> f16 per-WMMA-fragment layout.
// frag id = (i*KK + kk)*16 + ntile; within frag: lane*16 + h.
// B-operand (32x16): lane = n + 16*(k>=16), h = k&15.
// ---------------------------------------------------------------------------
__global__ void prep_w(const float* __restrict__ W, _Float16* __restrict__ Wp,
                       int Hk, int KK, int total) {
    int idx = blockIdx.x * blockDim.x + threadIdx.x;
    if (idx >= total) return;
    int h    = idx & 15;
    int lane = (idx >> 4) & 31;
    int frag = idx >> 9;
    int nt   = frag & 15;
    int ikk  = frag >> 4;
    int kk   = ikk % KK;
    int i    = ikk / KK;
    int n    = lane & 15;
    int kl   = ((lane >> 4) << 4) + h;
    int j    = kk * 32 + kl;
    _Float16 v = (_Float16)0.0f;
    if (j < Hk) v = (_Float16)W[(i * Hk + j) * UOUT + nt * 16 + n];
    Wp[idx] = v;
}

// ---------------------------------------------------------------------------
// Async (or fallback) staging of one 16-byte chunk global -> LDS.
// ---------------------------------------------------------------------------
__device__ __forceinline__ void stage_w16(const _Float16* src, _Float16* dst) {
#ifdef ASYNC_LDS_COPY
    __builtin_amdgcn_global_load_async_to_lds_b128(
        (__attribute__((address_space(1))) v4i_b128*)src,
        (__attribute__((address_space(3))) v4i_b128*)dst, 0, 0);
#else
    *(v8h*)dst = *(const v8h*)src;
#endif
}

template<int N>
__device__ __forceinline__ void wait_async_copies() {
#ifdef ASYNC_LDS_COPY
#if __has_builtin(__builtin_amdgcn_s_wait_asynccnt)
    __builtin_amdgcn_s_wait_asynccnt(N);
#else
    asm volatile("s_wait_asynccnt %0" :: "i"(N) : "memory");
#endif
#endif
}

// Scheduling fence: keep all B-fragment ds_loads issued before the WMMAs so
// the waitcnt pass can use partial s_wait_dscnt instead of 0 per WMMA.
__device__ __forceinline__ void sched_fence() {
#if defined(__gfx1250__) && __has_builtin(__builtin_amdgcn_sched_barrier)
    __builtin_amdgcn_sched_barrier(0);
#endif
}

// ---------------------------------------------------------------------------
// CIN layer, 512 threads = 16 waves: 8 row-blocks x 2 col-blocks.
// WG tile: 128 rows (4 batch elems) x 128 cols. A frags register-resident;
// per-step 8KB W tiles triple-buffered in LDS via async global->LDS copies
// issued 2 steps ahead (asynccnt<=1 across the barrier).
// ---------------------------------------------------------------------------
template<int KK, int HKP>
__global__ __launch_bounds__(512)
void cin_layer(const _Float16* __restrict__ xk,    // [MROWS, HKP] f16
               const _Float16* __restrict__ x0t,   // [MROWS, HKP0] f16 scales
               const _Float16* __restrict__ wp,    // swizzled weight fragments
               const float*    __restrict__ bias,  // [256]
               _Float16*       __restrict__ fout,  // [MROWS, 256] or nullptr
               float*          __restrict__ out,   // [BATCH, 768]
               int out_off) {
    __shared__ __align__(16) _Float16 wbuf[3 * WSTEP];  // 3 x 8KB W step tiles
    __shared__ float pool[4][NTILE];

    const int tid  = threadIdx.x;
    const int lane = tid & 31;
    const int wv   = tid >> 5;     // 0..15
    const int nw   = wv & 1;       // 64-col block within 128-col slice
    const int rw   = wv >> 1;      // 0..7 row block (16 rows each)
    const int m0   = blockIdx.x * MTILE;
    const int b0   = m0 >> 5;      // first batch element (4 per WG)
    const int ngrp = blockIdx.y;   // 0/1: which 128-col half

    ((float*)pool)[tid] = 0.0f;

    const int khalf = lane >> 4;
    const int mrow  = m0 + rw * 16 + (lane & 15);

    // Register-resident A fragments (contiguous k = 16*khalf + s, matching B).
    union AF { v16h v; v8h h[2]; };
    AF a[KK];
    const _Float16* xrow = xk + (size_t)mrow * HKP;
#pragma unroll
    for (int kk = 0; kk < KK; ++kk) {
        a[kk].h[0] = *(const v8h*)(xrow + kk * 32 + khalf * 16);
        a[kk].h[1] = *(const v8h*)(xrow + kk * 32 + khalf * 16 + 8);
    }

    v8f c[4];
#pragma unroll
    for (int t = 0; t < 4; ++t) { v8f z = {}; c[t] = z; }

    const _Float16* x0row = x0t + (size_t)mrow * HKP0;
    const int NSTEP = NF * KK;

    // Prologue: stage step 0 and step 1 tiles; step 0 must be complete.
    stage_w16(wp + ((size_t)0 * 16 + ngrp * 8) * 512 + (size_t)tid * 8,
              &wbuf[0 * WSTEP + tid * 8]);
    stage_w16(wp + ((size_t)1 * 16 + ngrp * 8) * 512 + (size_t)tid * 8,
              &wbuf[1 * WSTEP + tid * 8]);
    wait_async_copies<1>();
    __syncthreads();

    int step = 0;
    int cur = 0, nxt2 = 2;                 // step mod 3, (step+2) mod 3
    _Float16 s_pref = x0row[0];
    for (int i = 0; i < NF; ++i) {
        const _Float16 s = s_pref;
        if (i + 1 < NF) s_pref = x0row[i + 1];   // issue next scale load early
#pragma unroll
        for (int kk = 0; kk < KK; ++kk, ++step) {
            const bool fill = (step + 2 < NSTEP);
            if (fill) {   // stage step+2's tile 2 barriers ahead
                stage_w16(wp + ((size_t)(step + 2) * 16 + ngrp * 8) * 512 +
                              (size_t)tid * 8,
                          &wbuf[nxt2 * WSTEP + tid * 8]);
            }
            v16h as = a[kk].v * s;   // A = diag(x0[:,i]) * Xk  (packed f16)
            const _Float16* wb = &wbuf[cur * WSTEP + (nw * 4) * 512 + lane * 16];
            // Issue ALL 4 B-fragment loads, then fence, then the 4 WMMAs:
            // ds retires in order -> partial s_wait_dscnt (6/4/2/0).
            v16h bf0 = *(const v16h*)(wb);
            v16h bf1 = *(const v16h*)(wb + 512);
            v16h bf2 = *(const v16h*)(wb + 1024);
            v16h bf3 = *(const v16h*)(wb + 1536);
            sched_fence();
            c[0] = __builtin_amdgcn_wmma_f32_16x16x32_f16(
                false, as, false, bf0, (short)0, c[0], false, false);
            c[1] = __builtin_amdgcn_wmma_f32_16x16x32_f16(
                false, as, false, bf1, (short)0, c[1], false, false);
            c[2] = __builtin_amdgcn_wmma_f32_16x16x32_f16(
                false, as, false, bf2, (short)0, c[2], false, false);
            c[3] = __builtin_amdgcn_wmma_f32_16x16x32_f16(
                false, as, false, bf3, (short)0, c[3], false, false);
            // Newest fill may stay in flight across the barrier.
            if (fill) wait_async_copies<1>();
            else      wait_async_copies<0>();
            __syncthreads();
            cur  = (cur  == 2) ? 0 : cur + 1;
            nxt2 = (nxt2 == 2) ? 0 : nxt2 + 1;
        }
    }

    // bias add, fmap store (next layer's xk), and sum-over-D pooling
    const int b_loc = rw >> 1;     // which of the 4 batch elements
#pragma unroll
    for (int t = 0; t < 4; ++t) {
        const int n_loc = nw * 64 + t * 16 + (lane & 15);
        const int n     = ngrp * NTILE + n_loc;
        const float bv  = bias[n];
        float psum = 0.0f;
#pragma unroll
        for (int e = 0; e < 8; ++e) {   // C layout: VGPR e -> row e + 8*khalf
            float hval = c[t][e] + bv;
            psum += hval;
            if (fout) {
                const int mr = m0 + rw * 16 + khalf * 8 + e;
                fout[(size_t)mr * UOUT + n] = (_Float16)hval;
            }
        }
        atomicAdd(&pool[b_loc][n_loc], psum);  // ds_add_f32
    }
    __syncthreads();
    {
        const int bl = tid >> 7;           // 0..3
        const int nl = tid & (NTILE - 1);  // 0..127
        out[(size_t)(b0 + bl) * OUTW + out_off + ngrp * NTILE + nl] = pool[bl][nl];
    }
}

// ---------------------------------------------------------------------------
// Launcher. Input dict order: inputs, training, W0, b0, W1, b1, W2, b2.
// ---------------------------------------------------------------------------
extern "C" void kernel_launch(void* const* d_in, const int* in_sizes, int n_in,
                              void* d_out, int out_size, void* d_ws, size_t ws_size,
                              hipStream_t stream) {
    const float* inp = (const float*)d_in[0];
    const float* W0  = (const float*)d_in[2];
    const float* b0  = (const float*)d_in[3];
    const float* W1  = (const float*)d_in[4];
    const float* b1  = (const float*)d_in[5];
    const float* W2  = (const float*)d_in[6];
    const float* b2  = (const float*)d_in[7];
    float* out = (float*)d_out;

    char* ws = (char*)d_ws;
    size_t off = 0;
    auto alloc = [&](size_t bytes) -> void* {
        void* p = ws + off;
        off += (bytes + 255) & ~(size_t)255;
        return p;
    };
    _Float16* X0T = (_Float16*)alloc((size_t)MROWS * HKP0 * 2);
    _Float16* W0p = (_Float16*)alloc((size_t)NF * HKP0 * UOUT * 2);
    _Float16* W1p = (_Float16*)alloc((size_t)NF * UOUT * UOUT * 2);
    _Float16* W2p = (_Float16*)alloc((size_t)NF * UOUT * UOUT * 2);
    _Float16* F1  = (_Float16*)alloc((size_t)MROWS * UOUT * 2);
    _Float16* F2  = (_Float16*)alloc((size_t)MROWS * UOUT * 2);

    {
        int total = MROWS * HKP0;
        prep_x0<<<total / 256, 256, 0, stream>>>(inp, X0T);
    }
    {
        int total = NF * HKP0 * UOUT;   // layer 0: Hk=39 padded to 64, KK=2
        prep_w<<<(total + 255) / 256, 256, 0, stream>>>(W0, W0p, NF, 2, total);
    }
    {
        int total = NF * UOUT * UOUT;   // layers 1,2: Hk=256, KK=8
        prep_w<<<(total + 255) / 256, 256, 0, stream>>>(W1, W1p, UOUT, 8, total);
        prep_w<<<(total + 255) / 256, 256, 0, stream>>>(W2, W2p, UOUT, 8, total);
    }

    dim3 grid(MROWS / MTILE, UOUT / NTILE);   // (128, 2)
    cin_layer<2, HKP0><<<grid, 512, 0, stream>>>(X0T, X0T, W0p, b0, F1, out, 0);
    cin_layer<8, UOUT><<<grid, 512, 0, stream>>>(F1,  X0T, W1p, b1, F2, out, 256);
    cin_layer<8, UOUT><<<grid, 512, 0, stream>>>(F2,  X0T, W2p, b2,
                                                 (_Float16*)nullptr, out, 512);
}
